// GATEncoder_77171972374542
// MI455X (gfx1250) — compile-verified, matrix operand order checked by reading, not run on hardware
//
#include <hip/hip_runtime.h>
#include <hip/hip_bf16.h>
#include <stdint.h>

// ---------------------------------------------------------------------------
// GAT encoder for MI455X (gfx1250, wave32).
// Dense projections use v_wmma_f32_16x16x32_bf16 (bf16 in, fp32 accumulate).
// Edge softmax/scatter phases are HBM/atomic bound -> scalar VALU + atomics.
// ---------------------------------------------------------------------------

#define NNODES 50000
#define NEDGES 800000
#define INDIM  128
#define NHEADS 4
#define NEG_SLOPE 0.2f

typedef __attribute__((ext_vector_type(16))) __bf16 v16bf;
typedef __attribute__((ext_vector_type(8)))  float  v8f;

__device__ __forceinline__ __bf16 f2bf(float f) {
    union { float f; uint32_t u; } v; v.f = f;
    uint32_t r = v.u + 0x7FFFu + ((v.u >> 16) & 1u);   // round-to-nearest-even
    uint16_t h = (uint16_t)(r >> 16);
    __bf16 out;
    __builtin_memcpy(&out, &h, 2);
    return out;
}

// ---------------------------------------------------------------------------
// H[N x OD] = X[N x 128] @ W[128 x OD], bf16 WMMA, fp32 accum.
// Block = 256 threads = 8 waves, one 128-row slab per block.
// W is pre-swizzled into LDS in WMMA B-fragment order so each lane does
// contiguous 32B LDS reads per K-step. OD is compile-time (128 or 256).
// ---------------------------------------------------------------------------
template <int OD>
__global__ __launch_bounds__(256)
void gat_gemm_wmma(const float* __restrict__ X, const float* __restrict__ W,
                   float* __restrict__ H, int nrows) {
    __shared__ __bf16 wlds[INDIM * OD];

    const int tid = threadIdx.x;

    // ---- Stage W into LDS, swizzled to the WMMA 16-bit B-fragment layout.
    // slot s <-> (ntile, kstep kk, lane); each slot = 16 contiguous bf16.
    // ISA 16-bit layout: lanes 0-15 hold K = kk*32 + {0..7, 16..23},
    //                    lanes 16-31 hold K = kk*32 + {8..15, 24..31}.
    constexpr int SLOTS = OD * 8;                  // (OD/16) * 4 * 32
    for (int s = tid; s < SLOTS; s += 256) {
        const int ntile = s >> 7;                  // / 128
        const int rem   = s & 127;
        const int kk    = rem >> 5;
        const int lane  = rem & 31;
        const int n     = ntile * 16 + (lane & 15);
        const int half  = lane >> 4;
        const int base  = kk * 32 + half * 8;
        __bf16* dstp = &wlds[s * 16];
        #pragma unroll
        for (int i = 0; i < 8; ++i)
            dstp[i]     = f2bf(W[(base + i) * OD + n]);
        #pragma unroll
        for (int i = 0; i < 8; ++i)
            dstp[8 + i] = f2bf(W[(base + 16 + i) * OD + n]);
    }
    __syncthreads();

    const int wave = tid >> 5;
    const int lane = tid & 31;
    const int half = lane >> 4;
    const int row  = blockIdx.x * 128 + wave * 16 + (lane & 15);
    const bool rowok = (row < nrows);
    const bool block_full = (blockIdx.x * 128 + 128) <= nrows;   // uniform

    // ---- A fragments for all four K-steps (registers, reused across tiles).
    v16bf afrag[4];
    #pragma unroll
    for (int kk = 0; kk < 4; ++kk) {
        const int base = kk * 32 + half * 8;
        float4 f0 = make_float4(0.f, 0.f, 0.f, 0.f);
        float4 f1 = f0, f2 = f0, f3 = f0;
        if (rowok) {
            const float4* p = (const float4*)(X + (long)row * INDIM + base);
            f0 = p[0]; f1 = p[1];
            const float4* q = (const float4*)(X + (long)row * INDIM + base + 16);
            f2 = q[0]; f3 = q[1];
        }
        const float vals[16] = { f0.x, f0.y, f0.z, f0.w, f1.x, f1.y, f1.z, f1.w,
                                 f2.x, f2.y, f2.z, f2.w, f3.x, f3.y, f3.z, f3.w };
        #pragma unroll
        for (int i = 0; i < 16; ++i) afrag[kk][i] = f2bf(vals[i]);
    }

    // ---- Sweep column tiles: batch all 8 ds_loads, then 4 chained wmma.
    constexpr int NTILES = OD >> 4;
    const int rbase = blockIdx.x * 128 + wave * 16 + half * 8;
    const int col   = lane & 15;
    #pragma unroll 2
    for (int nt = 0; nt < NTILES; ++nt) {
        const v16bf* bbase = (const v16bf*)&wlds[(nt * 4 * 32 + lane) * 16];
        const v16bf b0 = bbase[0 * 32];
        const v16bf b1 = bbase[1 * 32];
        const v16bf b2 = bbase[2 * 32];
        const v16bf b3 = bbase[3 * 32];
        v8f acc = {0.f, 0.f, 0.f, 0.f, 0.f, 0.f, 0.f, 0.f};
        acc = __builtin_amdgcn_wmma_f32_16x16x32_bf16(false, afrag[0], false, b0, (short)0, acc, false, false);
        acc = __builtin_amdgcn_wmma_f32_16x16x32_bf16(false, afrag[1], false, b1, (short)0, acc, false, false);
        acc = __builtin_amdgcn_wmma_f32_16x16x32_bf16(false, afrag[2], false, b2, (short)0, acc, false, false);
        acc = __builtin_amdgcn_wmma_f32_16x16x32_bf16(false, afrag[3], false, b3, (short)0, acc, false, false);

        // D layout: VGPR r -> (M = r + 8*half, N = lane&15).
        float* bp = H + (long)rbase * OD + nt * 16 + col;
        if (block_full) {                 // uniform scalar branch, no exec churn
            #pragma unroll
            for (int r = 0; r < 8; ++r) bp[(long)r * OD] = acc[r];
        } else {
            #pragma unroll
            for (int r = 0; r < 8; ++r)
                if (rbase + r < nrows) bp[(long)r * OD] = acc[r];
        }
    }
}

// alpha_src/alpha_dst logits: one thread per (node, head).
__global__ __launch_bounds__(256)
void alpha_kernel(const float* __restrict__ H, const float* __restrict__ a_src,
                  const float* __restrict__ a_dst, float* __restrict__ as_o,
                  float* __restrict__ ad_o, int n_nodes, int C) {
    const int t = blockIdx.x * blockDim.x + threadIdx.x;
    if (t >= n_nodes * NHEADS) return;
    const int n = t >> 2, hh = t & 3;
    const float* hp  = H + (long)n * NHEADS * C + hh * C;
    const float* asv = a_src + hh * C;
    const float* adv = a_dst + hh * C;
    float s1 = 0.f, s2 = 0.f;
    for (int c = 0; c < C; ++c) { const float v = hp[c]; s1 += v * asv[c]; s2 += v * adv[c]; }
    as_o[t] = s1;
    ad_o[t] = s2;
}

__global__ __launch_bounds__(256)
void fill_kernel(uint32_t* __restrict__ p, long count, uint32_t bits) {
    const long t = blockIdx.x * (long)blockDim.x + threadIdx.x;
    if (t < count) p[t] = bits;
}

__device__ __forceinline__ float leaky(float x) {
    return x > 0.f ? x : NEG_SLOPE * x;
}

// segment max of leaky_relu(alpha_src[src] + alpha_dst[dst]) over dst.
// float atomic-max via signed-max / unsigned-min bit trick (init = -inf).
__global__ __launch_bounds__(256)
void edge_max_kernel(const int* __restrict__ src, const int* __restrict__ dst,
                     const float* __restrict__ as_o, const float* __restrict__ ad_o,
                     float* __restrict__ m) {
    const int t = blockIdx.x * blockDim.x + threadIdx.x;
    if (t >= NEDGES * NHEADS) return;
    const int e = t >> 2, hh = t & 3;
    const int s = src[e], d = dst[e];
    const float x = leaky(as_o[s * NHEADS + hh] + ad_o[d * NHEADS + hh]);
    float* addr = &m[d * NHEADS + hh];
    if (x >= 0.f) atomicMax((int*)addr, __float_as_int(x));
    else          atomicMin((unsigned int*)addr, __float_as_uint(x));
}

// denom[dst,h] += exp(e - m[dst,h])
__global__ __launch_bounds__(256)
void edge_sum_kernel(const int* __restrict__ src, const int* __restrict__ dst,
                     const float* __restrict__ as_o, const float* __restrict__ ad_o,
                     const float* __restrict__ m, float* __restrict__ den) {
    const int t = blockIdx.x * blockDim.x + threadIdx.x;
    if (t >= NEDGES * NHEADS) return;
    const int e = t >> 2, hh = t & 3;
    const int s = src[e], d = dst[e];
    const float x = leaky(as_o[s * NHEADS + hh] + ad_o[d * NHEADS + hh]);
    atomicAdd(&den[d * NHEADS + hh], __expf(x - m[d * NHEADS + hh]));
}

// accum[dst, h, :] += alpha * H[src, h, :]
__global__ __launch_bounds__(256)
void edge_scatter_kernel(const int* __restrict__ src, const int* __restrict__ dst,
                         const float* __restrict__ as_o, const float* __restrict__ ad_o,
                         const float* __restrict__ m, const float* __restrict__ den,
                         const float* __restrict__ H, float* __restrict__ accum, int C) {
    const int t = blockIdx.x * blockDim.x + threadIdx.x;
    if (t >= NEDGES * NHEADS) return;
    const int e = t >> 2, hh = t & 3;
    const int s = src[e], d = dst[e];
    const float x = leaky(as_o[s * NHEADS + hh] + ad_o[d * NHEADS + hh]);
    const float alpha = __expf(x - m[d * NHEADS + hh]) / (den[d * NHEADS + hh] + 1e-16f);
    const float4* hp = (const float4*)(H + (long)s * NHEADS * C + hh * C);
    float* op = accum + (long)d * NHEADS * C + hh * C;
    for (int c = 0; c < (C >> 2); ++c) {
        const float4 v = hp[c];
        atomicAdd(op + 4 * c + 0, alpha * v.x);
        atomicAdd(op + 4 * c + 1, alpha * v.y);
        atomicAdd(op + 4 * c + 2, alpha * v.z);
        atomicAdd(op + 4 * c + 3, alpha * v.w);
    }
}

// concat layers: x_next = accum + bias
__global__ __launch_bounds__(256)
void add_bias_kernel(const float* __restrict__ accum, const float* __restrict__ bias,
                     float* __restrict__ xout, long n_elems, int dim) {
    const long t = blockIdx.x * (long)blockDim.x + threadIdx.x;
    if (t >= n_elems) return;
    xout[t] = accum[t] + bias[t % dim];
}

// final layer: mean over heads + bias
__global__ __launch_bounds__(256)
void mean_heads_kernel(const float* __restrict__ accum, const float* __restrict__ bias,
                       float* __restrict__ out, int n_nodes) {
    const int t = blockIdx.x * blockDim.x + threadIdx.x;
    if (t >= n_nodes * 64) return;
    const int n = t >> 6, c = t & 63;
    const float* p = accum + (long)n * 256;
    out[t] = 0.25f * (p[c] + p[64 + c] + p[128 + c] + p[192 + c]) + bias[c];
}

// ---------------------------------------------------------------------------

extern "C" void kernel_launch(void* const* d_in, const int* in_sizes, int n_in,
                              void* d_out, int out_size, void* d_ws, size_t ws_size,
                              hipStream_t stream) {
    (void)in_sizes; (void)n_in; (void)out_size; (void)ws_size;

    const float* x    = (const float*)d_in[0];
    const int*   ei   = (const int*)d_in[1];
    const int*   srcp = ei;
    const int*   dstp = ei + NEDGES;
    const float* W[3]     = { (const float*)d_in[2],  (const float*)d_in[6],  (const float*)d_in[10] };
    const float* a_src[3] = { (const float*)d_in[3],  (const float*)d_in[7],  (const float*)d_in[11] };
    const float* a_dst[3] = { (const float*)d_in[4],  (const float*)d_in[8],  (const float*)d_in[12] };
    const float* bias[3]  = { (const float*)d_in[5],  (const float*)d_in[9],  (const float*)d_in[13] };

    char* ws = (char*)d_ws;
    float* h_buf = (float*)(ws);                            // N*256 f32
    float* pp0   = (float*)(ws + (size_t)51200000);         // N*256 f32
    float* pp1   = (float*)(ws + (size_t)102400000);        // N*128 f32
    float* as_b  = (float*)(ws + (size_t)128000000);        // N*4
    float* ad_b  = as_b + (size_t)NNODES * NHEADS;
    float* m_b   = ad_b + (size_t)NNODES * NHEADS;
    float* den_b = m_b  + (size_t)NNODES * NHEADS;

    const int GEMM_BLOCKS = (NNODES + 127) / 128;           // 391
    const int NH          = NNODES * NHEADS;
    const int NH_BLOCKS   = (NH + 255) / 256;
    const int EH          = NEDGES * NHEADS;
    const int EH_BLOCKS   = (EH + 255) / 256;

    const int   outd[3] = { 128, 128, 256 };
    const int   chan[3] = { 32, 32, 64 };
    const float* layer_in[3] = { x, pp1, pp1 };

    for (int L = 0; L < 3; ++L) {
        const int od = outd[L], C = chan[L];
        const float* xin = layer_in[L];
        float* acc = pp0;

        // 1) projection h = xin @ W[L]   (WMMA bf16)
        if (od == 128)
            gat_gemm_wmma<128><<<GEMM_BLOCKS, 256, 0, stream>>>(xin, W[L], h_buf, NNODES);
        else
            gat_gemm_wmma<256><<<GEMM_BLOCKS, 256, 0, stream>>>(xin, W[L], h_buf, NNODES);

        // 2) attention logits per (node, head)
        alpha_kernel<<<NH_BLOCKS, 256, 0, stream>>>(h_buf, a_src[L], a_dst[L],
                                                    as_b, ad_b, NNODES, C);

        // 3) init m = -inf, denom = 0, accum = 0
        fill_kernel<<<NH_BLOCKS, 256, 0, stream>>>((uint32_t*)m_b,  (long)NH, 0xFF800000u);
        fill_kernel<<<NH_BLOCKS, 256, 0, stream>>>((uint32_t*)den_b,(long)NH, 0u);
        {
            const long cnt = (long)NNODES * od;
            fill_kernel<<<(int)((cnt + 255) / 256), 256, 0, stream>>>((uint32_t*)acc, cnt, 0u);
        }

        // 4) segment softmax + weighted scatter over edges
        edge_max_kernel<<<EH_BLOCKS, 256, 0, stream>>>(srcp, dstp, as_b, ad_b, m_b);
        edge_sum_kernel<<<EH_BLOCKS, 256, 0, stream>>>(srcp, dstp, as_b, ad_b, m_b, den_b);
        edge_scatter_kernel<<<EH_BLOCKS, 256, 0, stream>>>(srcp, dstp, as_b, ad_b,
                                                           m_b, den_b, h_buf, acc, C);

        // 5) finalize
        if (L < 2) {
            const long cnt = (long)NNODES * od;
            add_bias_kernel<<<(int)((cnt + 255) / 256), 256, 0, stream>>>(
                acc, bias[L], pp1, cnt, od);
        } else {
            mean_heads_kernel<<<(NNODES * 64 + 255) / 256, 256, 0, stream>>>(
                acc, bias[L], (float*)d_out, NNODES);
        }
    }
}
